// WindowAttention_83502754169476
// MI455X (gfx1250) — matrix operand order repaired and checked by reading, not compiled, not run on hardware
//
#include <hip/hip_runtime.h>

// ---------------- types / helpers ----------------
typedef __bf16 v16bf __attribute__((ext_vector_type(16)));
typedef float  v8f   __attribute__((ext_vector_type(8)));

union BF16x16 {
    v16bf          v;
    uint4          q[2];
    unsigned short us[16];
};

__device__ __forceinline__ unsigned short f2bf(float f) {
    unsigned int u = __float_as_uint(f);
    u += 0x7FFFu + ((u >> 16) & 1u);   // round-to-nearest-even
    return (unsigned short)(u >> 16);
}

__device__ __forceinline__ v8f vzero8() {
    v8f z;
#pragma unroll
    for (int i = 0; i < 8; ++i) z[i] = 0.f;
    return z;
}

#define WMMA_BF16(A, B, C) \
    __builtin_amdgcn_wmma_f32_16x16x32_bf16(false, (A), false, (B), (short)0, (C), false, false)

// ds_swizzle group-of-32 xor reductions (and=0x1f, or=0, xor in bits [14:10])
__device__ __forceinline__ float xor_max16(float v) {
    v = fmaxf(v, __int_as_float(__builtin_amdgcn_ds_swizzle(__float_as_int(v), 0x041f))); // xor 1
    v = fmaxf(v, __int_as_float(__builtin_amdgcn_ds_swizzle(__float_as_int(v), 0x081f))); // xor 2
    v = fmaxf(v, __int_as_float(__builtin_amdgcn_ds_swizzle(__float_as_int(v), 0x101f))); // xor 4
    v = fmaxf(v, __int_as_float(__builtin_amdgcn_ds_swizzle(__float_as_int(v), 0x201f))); // xor 8
    return v;
}
__device__ __forceinline__ float xor_sum16(float v) {
    v += __int_as_float(__builtin_amdgcn_ds_swizzle(__float_as_int(v), 0x041f));
    v += __int_as_float(__builtin_amdgcn_ds_swizzle(__float_as_int(v), 0x081f));
    v += __int_as_float(__builtin_amdgcn_ds_swizzle(__float_as_int(v), 0x101f));
    v += __int_as_float(__builtin_amdgcn_ds_swizzle(__float_as_int(v), 0x201f));
    return v;
}

// ---------------- weight transpose + bf16 convert ----------------
// W: R x C (row-major f32) -> Wt: C x R (row-major bf16), Wt[c][r] = W[r][c]
__global__ void wa_transpose_bf16(const float* __restrict__ W,
                                  unsigned short* __restrict__ Wt,
                                  int R, int C) {
    int idx = blockIdx.x * 256 + threadIdx.x;
    if (idx < R * C) {
        int r = idx / C;
        int c = idx - r * C;
        Wt[(size_t)c * R + r] = f2bf(W[(size_t)r * C + c]);
    }
}

// ---------------- bf16-WMMA GEMM, single-pass over A ----------------
// C[M,N] = A[M,K(=384)] * Bt[N,K]^T   (Bt row n = column n of B, contiguous in K)
// STAGE 0: A f32 (convert on stage), C bf16, no bias
// STAGE 1: A bf16 (async global->LDS copy), C f32 + bias[n]
// WG = 128 threads (4 waves). Per WG: 64-row A panel staged ONCE into LDS,
// then all N column blocks (128 wide) are computed from it -> A read once from HBM.
#define AK   384
#define APAD 392   // row stride in elements: 784B = 4 banks mod 64 -> conflict-free

template <int STAGE>
__global__ __launch_bounds__(128) void wa_gemm_wmma(
    const void* __restrict__ Ap, const unsigned short* __restrict__ Bt,
    const float* __restrict__ bias, void* __restrict__ Cp,
    int M, int N) {
    __shared__ unsigned short As[64 * APAD];   // ~49 KB

    const int tid = threadIdx.x;
    const int row0 = blockIdx.x * 64;

    // ---- stage 64 x 384 A panel into LDS (bf16)
    if (STAGE == 0) {
        const float* A = (const float*)Ap;
        // 64*384/4 = 6144 float4 chunks, 48 per thread (f32 -> bf16 convert)
        for (int i = tid; i < 6144; i += 128) {
            const int r  = i / 96;          // 96 float4 per row
            const int c4 = i - r * 96;
            float4 f = *(const float4*)(A + (size_t)(row0 + r) * AK + c4 * 4);
            unsigned short h[4] = {f2bf(f.x), f2bf(f.y), f2bf(f.z), f2bf(f.w)};
            *(uint2*)&As[r * APAD + c4 * 4] = *(const uint2*)h;
        }
        __syncthreads();
    } else {
        const unsigned short* A = (const unsigned short*)Ap;
        // pure byte-copy: use CDNA5 async global->LDS (per-lane LDS dest handles APAD)
        // 64*384/8 = 3072 16-byte chunks, 24 per thread, all lanes active.
        for (int i = tid; i < 3072; i += 128) {
            const int r  = i / 48;          // 48 x (8 bf16) per row
            const int c8 = i - r * 48;
            unsigned lds_off = (unsigned)(size_t)(&As[r * APAD + c8 * 8]);
            unsigned long long ga =
                (unsigned long long)(A + (size_t)(row0 + r) * AK + c8 * 8);
            asm volatile("global_load_async_to_lds_b128 %0, %1, off"
                         :: "v"(lds_off), "v"(ga) : "memory");
        }
        asm volatile("s_wait_asynccnt 0" ::: "memory");
        __syncthreads();
    }

    const int wv = tid >> 5;
    const int L  = tid & 31;
    const int lh = L & 15;   // lane within 16-half
    const int hi = L >> 4;   // which half
    const int mrow = wv * 16 + lh;   // A-fragment row within panel

    for (int col0 = 0; col0 < N; col0 += 128) {
        v8f acc[8];
#pragma unroll
        for (int j = 0; j < 8; ++j) acc[j] = vzero8();

#pragma unroll
        for (int k0 = 0; k0 < AK; k0 += 32) {
            // A fragment: VGPR0-3: k = k0+hi*8..+7 ; VGPR4-7: k = k0+16+hi*8..+7
            BF16x16 a;
            a.q[0] = *(const uint4*)&As[mrow * APAD + k0 + hi * 8];
            a.q[1] = *(const uint4*)&As[mrow * APAD + k0 + 16 + hi * 8];
#pragma unroll
            for (int j = 0; j < 8; ++j) {
                const int bn = col0 + j * 16 + lh;
                const unsigned short* brow = Bt + (size_t)bn * AK + k0 + hi * 8;
                BF16x16 b;
                b.q[0] = ((const uint4*)brow)[0];
                b.q[1] = ((const uint4*)(brow + 16))[0];
                acc[j] = WMMA_BF16(a.v, b.v, acc[j]);
            }
        }

        // C layout: lane(0-15)=N, VGPR r -> M=r (+8 for upper half)
#pragma unroll
        for (int j = 0; j < 8; ++j) {
#pragma unroll
            for (int r = 0; r < 8; ++r) {
                const int m = row0 + wv * 16 + r + 8 * hi;
                const int n = col0 + j * 16 + lh;
                const float val = acc[j][r];
                if (STAGE == 0)
                    ((unsigned short*)Cp)[(size_t)m * N + n] = f2bf(val);
                else
                    ((float*)Cp)[(size_t)m * N + n] = val + bias[n];
            }
        }
    }
}

// ---------------- window attention core ----------------
// One workgroup per (batch, head, window). N=49 tokens padded to 64, d=32.
#define QKPAD 40   // 80B  = 20 banks mod 64 -> conflict-free
#define VPAD  72   // 144B = 36 banks mod 64 -> conflict-free
__global__ __launch_bounds__(128) void wa_window_attn(
    const unsigned short* __restrict__ qkv,
    const float* __restrict__ bias_table,
    unsigned short* __restrict__ attn_out) {
    __shared__ unsigned short Qs[64 * QKPAD];
    __shared__ unsigned short Ks[64 * QKPAD];
    __shared__ unsigned short Vt[32 * VPAD];   // transposed V: Vt[d][token]
    __shared__ unsigned short Sa[64 * VPAD];   // post-softmax attention, bf16

    const int bid  = blockIdx.x;
    const int head = bid % 12;
    const int win  = (bid / 12) & 63;
    const int b    = bid / (12 * 64);
    const int wy = win >> 3, wx = win & 7;
    const int tid = threadIdx.x;

    // ---- stage Q, K, V^T into LDS (rows/cols >= 49 zero-padded)
    for (int e = tid; e < 2048; e += 128) {
        const int n  = e >> 5;
        const int dd = e & 31;
        unsigned short q = 0, k = 0, v = 0;
        if (n < 49) {
            const int ty = n / 7, tx = n - (n / 7) * 7;
            const size_t rowi = ((size_t)(b * 56 + wy * 7 + ty)) * 56 + (wx * 7 + tx);
            const unsigned short* p = qkv + rowi * 1152 + head * 32 + dd;
            q = p[0];
            k = p[384];
            v = p[768];
        }
        Qs[n * QKPAD + dd] = q;
        Ks[n * QKPAD + dd] = k;
        Vt[dd * VPAD + n]  = v;
    }
    __syncthreads();

    const int wv = tid >> 5, L = tid & 31, lh = L & 15, hi = L >> 4;
    const int m0 = wv * 16;   // this wave's row strip

    // ---- S = Q * K^T * scale  (16 rows x 64 cols = 4 WMMA tiles per wave)
    BF16x16 aq;
    aq.q[0] = *(const uint4*)&Qs[(m0 + lh) * QKPAD + hi * 8];
    aq.q[1] = *(const uint4*)&Qs[(m0 + lh) * QKPAD + 16 + hi * 8];

    float s[4][8];
#pragma unroll
    for (int j = 0; j < 4; ++j) {
        const unsigned short* p = &Ks[(j * 16 + lh) * QKPAD + hi * 8];
        BF16x16 bk;
        bk.q[0] = ((const uint4*)p)[0];
        bk.q[1] = ((const uint4*)(p + 16))[0];
        v8f c = vzero8();
        c = WMMA_BF16(aq.v, bk.v, c);
#pragma unroll
        for (int r = 0; r < 8; ++r) {
            float t = c[r] * 0.17677669529663687f;          // 1/sqrt(32)
            if (j * 16 + lh >= 49) t = -__builtin_inff();   // mask padded keys
            s[j][r] = t;
        }
    }

    // ---- row softmax via ds_swizzle xor reductions (rows distinct per 16-lane half),
    //      then bias added AFTER softmax (faithful to reference)
#pragma unroll
    for (int r = 0; r < 8; ++r) {
        float mx = xor_max16(fmaxf(fmaxf(s[0][r], s[1][r]), fmaxf(s[2][r], s[3][r])));
        float sum = 0.f;
#pragma unroll
        for (int j = 0; j < 4; ++j) {
            const float e = __expf(s[j][r] - mx);
            s[j][r] = e;
            sum += e;
        }
        sum = xor_sum16(sum);
        const float inv = 1.0f / sum;

        const int m = m0 + r + 8 * hi;
#pragma unroll
        for (int j = 0; j < 4; ++j) {
            float a = s[j][r] * inv;
            const int col = j * 16 + lh;
            if (m < 49 && col < 49) {
                const int ry = m / 7,   rx = m - ry * 7;
                const int cy = col / 7, cx = col - cy * 7;
                const int idx = (ry - cy + 6) * 13 + (rx - cx + 6);
                a += bias_table[idx * 12 + head];
            }
            Sa[m * VPAD + col] = f2bf(a);   // padded cols stay 0; V pad rows are 0 anyway
        }
    }
    __syncthreads();

    // ---- O = A * V  (16x64 times 64x32, K split into 2 WMMA per d-tile)
    v8f o0 = vzero8(), o1 = vzero8();
#pragma unroll
    for (int kb = 0; kb < 2; ++kb) {
        BF16x16 aa;
        aa.q[0] = *(const uint4*)&Sa[(m0 + lh) * VPAD + kb * 32 + hi * 8];
        aa.q[1] = *(const uint4*)&Sa[(m0 + lh) * VPAD + kb * 32 + 16 + hi * 8];
        const unsigned short* p0 = &Vt[(0 + lh) * VPAD + kb * 32 + hi * 8];
        const unsigned short* p1 = &Vt[(16 + lh) * VPAD + kb * 32 + hi * 8];
        BF16x16 b0, b1;
        b0.q[0] = ((const uint4*)p0)[0]; b0.q[1] = ((const uint4*)(p0 + 16))[0];
        b1.q[0] = ((const uint4*)p1)[0]; b1.q[1] = ((const uint4*)(p1 + 16))[0];
        o0 = WMMA_BF16(aa.v, b0.v, o0);
        o1 = WMMA_BF16(aa.v, b1.v, o1);
    }

    // ---- scatter valid rows back to (b, y, x, head*32+d) bf16
#pragma unroll
    for (int r = 0; r < 8; ++r) {
        const int m = m0 + r + 8 * hi;
        if (m < 49) {
            const int ty = m / 7, tx = m - (m / 7) * 7;
            const size_t rowi = ((size_t)(b * 56 + wy * 7 + ty)) * 56 + (wx * 7 + tx);
            unsigned short* dst = attn_out + rowi * 384 + head * 32;
            dst[lh]      = f2bf(o0[r]);
            dst[16 + lh] = f2bf(o1[r]);
        }
    }
}

// ---------------- launcher ----------------
extern "C" void kernel_launch(void* const* d_in, const int* in_sizes, int n_in,
                              void* d_out, int out_size, void* d_ws, size_t ws_size,
                              hipStream_t stream) {
    (void)in_sizes; (void)n_in; (void)out_size; (void)ws_size;
    const float* x          = (const float*)d_in[0];   // (32,56,56,384)
    const float* w_qkv      = (const float*)d_in[1];   // (384,1152)
    const float* w_out      = (const float*)d_in[2];   // (384,384)
    const float* b_out      = (const float*)d_in[3];   // (384,)
    const float* bias_table = (const float*)d_in[4];   // (169,12)
    float* out = (float*)d_out;

    const int M = 32 * 56 * 56;   // 100352
    const int H = 384, NQKV = 1152;

    char* ws = (char*)d_ws;
    size_t off = 0;
    unsigned short* qkv   = (unsigned short*)(ws + off); off += (size_t)M * NQKV * 2;  // 220.5 MB
    unsigned short* attnb = (unsigned short*)(ws + off); off += (size_t)M * H * 2;     // 73.5 MB
    unsigned short* WtQ   = (unsigned short*)(ws + off); off += (size_t)NQKV * H * 2;
    unsigned short* WtO   = (unsigned short*)(ws + off); off += (size_t)H * H * 2;

    // 1-2. weight transpose + bf16 convert (weights stay L2-resident afterwards)
    wa_transpose_bf16<<<(H * NQKV + 255) / 256, 256, 0, stream>>>(w_qkv, WtQ, H, NQKV);
    wa_transpose_bf16<<<(H * H + 255) / 256, 256, 0, stream>>>(w_out, WtO, H, H);

    // 3. qkv = x @ w_qkv  (f32 in, bf16 out) -- x read from HBM exactly once
    wa_gemm_wmma<0><<<M / 64, 128, 0, stream>>>(
        (const void*)x, WtQ, nullptr, (void*)qkv, M, NQKV);

    // 4. windowed attention (32 batches * 64 windows * 12 heads)
    wa_window_attn<<<32 * 64 * 12, 128, 0, stream>>>(qkv, bias_table, attnb);

    // 5. out = attn @ w_out + b_out  (bf16 in, f32 out, async A-panel staging)
    wa_gemm_wmma<1><<<M / 64, 128, 0, stream>>>(
        (const void*)attnb, WtO, b_out, (void*)out, M, H);
}
